// DKT_85727547228547
// MI455X (gfx1250) — matrix operand order repaired and checked by reading, not compiled, not run on hardware
//
#include <hip/hip_runtime.h>
#include <hip/hip_bf16.h>

typedef _Float16 h16;
typedef __attribute__((ext_vector_type(4)))  _Float16 v4h;
typedef __attribute__((ext_vector_type(16))) _Float16 v16h;
typedef __attribute__((ext_vector_type(4)))  float    v4f;
typedef __attribute__((ext_vector_type(8)))  float    v8f;

#define DEVINL __device__ __forceinline__

// ---------------------------------------------------------------------------
// WMMA helpers (this code's wave32 layouts, V_WMMA_F32_16X16X32_F16)
// A (16x32 f16, row-major src [M][K]):
//   lanes 0-15 : M=lane,    halves 0..15 = K k0..k0+15
//   lanes 16-31: M=lane-16, halves 0..15 = K k0+16..k0+31
// B (32x16 f16) loaded from Bt = B^T row-major [N][K]:
//   n = lane&15, half j = K = k0 + j + (lane>=16 ? 16 : 0)  (16 contiguous K)
// C/D (16x16 f32): reg r -> M = m0 + r + (lane>=16 ? 8 : 0), N = n0 + (lane&15)
// ---------------------------------------------------------------------------

DEVINL v8f wmma_f16(v16h a, v16h b, v8f c) {
  return __builtin_amdgcn_wmma_f32_16x16x32_f16(
      /*neg_a=*/false, a, /*neg_b=*/false, b,
      /*c_mod=*/(short)0, c, /*reuse_a=*/false, /*reuse_b=*/false);
}

// Unguarded fast loaders (all indices known in-range; EXEC stays full)
DEVINL v16h load_a_nc(const h16* __restrict__ A, int lda, int m0, int k0,
                      int lane) {
  int m   = m0 + (lane & 15);
  int klo = k0 + ((lane & 16) ? 16 : 0);
  const h16* row = A + (size_t)m * lda;
  const v4h* p0 = (const v4h*)(row + klo);
  const v4h* p1 = (const v4h*)(row + klo + 8);
  v4h x0 = p0[0], x1 = p0[1], x2 = p1[0], x3 = p1[1];
  v16h a;
#pragma unroll
  for (int j = 0; j < 4; ++j) {
    a[j] = x0[j]; a[4 + j] = x1[j]; a[8 + j] = x2[j]; a[12 + j] = x3[j];
  }
  return a;
}

DEVINL v16h load_b_nc(const h16* __restrict__ Bt, int ldb, int n0, int k0,
                      int lane) {
  int n  = n0 + (lane & 15);
  int kb = k0 + ((lane & 16) ? 16 : 0);
  const v4h* p = (const v4h*)(Bt + (size_t)n * ldb + kb);
  v4h x0 = p[0], x1 = p[1], x2 = p[2], x3 = p[3];
  v16h b;
#pragma unroll
  for (int j = 0; j < 4; ++j) {
    b[j] = x0[j]; b[4 + j] = x1[j]; b[8 + j] = x2[j]; b[12 + j] = x3[j];
  }
  return b;
}

// Guarded loaders (edge tiles only)
DEVINL v16h load_a_frag(const h16* __restrict__ A, int M, int K, int lda,
                        int m0, int k0, int lane) {
  v16h a = {};
  int m   = m0 + (lane & 15);
  int klo = k0 + ((lane & 16) ? 16 : 0);
  int khi = klo + 8;
  if (m < M) {
    const h16* row = A + (size_t)m * lda;
    if (khi + 7 < K) {
      const v4h* p0 = (const v4h*)(row + klo);
      const v4h* p1 = (const v4h*)(row + khi);
      v4h x0 = p0[0], x1 = p0[1], x2 = p1[0], x3 = p1[1];
#pragma unroll
      for (int j = 0; j < 4; ++j) {
        a[j] = x0[j]; a[4 + j] = x1[j]; a[8 + j] = x2[j]; a[12 + j] = x3[j];
      }
    } else {
#pragma unroll
      for (int j = 0; j < 8; ++j) {
        int ka = klo + j, kb = khi + j;
        a[j]     = (ka < K) ? row[ka] : (h16)0.f;
        a[8 + j] = (kb < K) ? row[kb] : (h16)0.f;
      }
    }
  }
  return a;
}

DEVINL v16h load_b_frag(const h16* __restrict__ Bt, int N, int K, int ldb,
                        int n0, int k0, int lane) {
  v16h b = {};
  int n  = n0 + (lane & 15);
  int kb = k0 + ((lane & 16) ? 16 : 0);
  if (n < N) {
    const h16* row = Bt + (size_t)n * ldb;
    if (kb + 15 < K) {
      const v4h* p = (const v4h*)(row + kb);
      v4h x0 = p[0], x1 = p[1], x2 = p[2], x3 = p[3];
#pragma unroll
      for (int j = 0; j < 4; ++j) {
        b[j] = x0[j]; b[4 + j] = x1[j]; b[8 + j] = x2[j]; b[12 + j] = x3[j];
      }
    } else {
#pragma unroll
      for (int j = 0; j < 16; ++j) {
        int k = kb + j;
        b[j] = (k < K) ? row[k] : (h16)0.f;
      }
    }
  }
  return b;
}

DEVINL void store_c_f32(float* __restrict__ C, int M, int N, int ldc,
                        int m0, int n0, int lane, v8f acc, const float* bias) {
  int n = n0 + (lane & 15);
  if (n >= N) return;
  float bv = bias ? bias[n] : 0.f;
  int mb = m0 + ((lane & 16) ? 8 : 0);
#pragma unroll
  for (int r = 0; r < 8; ++r) {
    int m = mb + r;
    if (m < M) C[(size_t)m * ldc + n] = acc[r] + bv;
  }
}

DEVINL void store_c_f16(h16* __restrict__ C, int M, int N, int ldc,
                        int m0, int n0, int lane, v8f acc, const float* bias) {
  int n = n0 + (lane & 15);
  if (n >= N) return;
  float bv = bias ? bias[n] : 0.f;
  int mb = m0 + ((lane & 16) ? 8 : 0);
#pragma unroll
  for (int r = 0; r < 8; ++r) {
    int m = mb + r;
    if (m < M) C[(size_t)m * ldc + n] = (h16)(acc[r] + bv);
  }
}

// ---------------------------------------------------------------------------
// Generic WMMA GEMM: C[M,N] = A[M,K] * Bt[N,K]^T (+bias), optional batching.
// Block = 256 threads = 8 waves; each wave: 1 M-tile x 4 N-tiles.
// Guards are selected by UNIFORM conditions; K-tail peeled.
// ---------------------------------------------------------------------------
template <int CT>
__global__ void gemm_wmma_k(const h16* __restrict__ A, const h16* __restrict__ Bt,
                            void* __restrict__ Cv, const float* __restrict__ bias,
                            int M, int N, int K, int lda, int ldb, int ldc,
                            long sA, long sB, long sC) {
  int lane = threadIdx.x & 31;
  int wave = threadIdx.x >> 5;
  int z = blockIdx.z;
  A  += (size_t)z * (size_t)sA;
  Bt += (size_t)z * (size_t)sB;
  int m0 = blockIdx.x * 16;
  bool aFull = (m0 + 16 <= M);
  int ntbase = (blockIdx.y * 8 + wave) * 4;  // first N-tile of this wave
  v8f acc[4] = {};
  int kFull = K >> 5;
  for (int kt = 0; kt < kFull; ++kt) {
    int k0 = kt << 5;
    v16h a = aFull ? load_a_nc(A, lda, m0, k0, lane)
                   : load_a_frag(A, M, K, lda, m0, k0, lane);
    if (kt + 1 < kFull)  // global_prefetch_b8 on the weight stream
      __builtin_prefetch(Bt + (size_t)((ntbase << 4) + (lane & 15)) * ldb + k0 + 32, 0, 1);
#pragma unroll
    for (int t = 0; t < 4; ++t) {
      int n0 = (ntbase + t) << 4;
      if (n0 + 16 <= N) {
        v16h b = load_b_nc(Bt, ldb, n0, k0, lane);
        acc[t] = wmma_f16(a, b, acc[t]);
      } else if (n0 < N) {
        v16h b = load_b_frag(Bt, N, K, ldb, n0, k0, lane);
        acc[t] = wmma_f16(a, b, acc[t]);
      }
    }
  }
  if (K & 31) {  // K tail (guarded)
    int k0 = kFull << 5;
    v16h a = load_a_frag(A, M, K, lda, m0, k0, lane);
#pragma unroll
    for (int t = 0; t < 4; ++t) {
      int n0 = (ntbase + t) << 4;
      if (n0 < N) {
        v16h b = load_b_frag(Bt, N, K, ldb, n0, k0, lane);
        acc[t] = wmma_f16(a, b, acc[t]);
      }
    }
  }
#pragma unroll
  for (int t = 0; t < 4; ++t) {
    int n0 = (ntbase + t) << 4;
    if (n0 >= N) continue;
    if (CT == 0) {
      float* C = (float*)Cv + (size_t)z * (size_t)sC;
      store_c_f32(C, M, N, ldc, m0, n0, lane, acc[t], bias);
    } else {
      h16* C = (h16*)Cv + (size_t)z * (size_t)sC;
      store_c_f16(C, M, N, ldc, m0, n0, lane, acc[t], bias);
    }
  }
}

// ---------------------------------------------------------------------------
// Embedding GEMM: xh/xd[16000,256] = x[16000,6000](f32) * quesT[256,6000]^T
// Block = 256 thr (8 waves) handles 32 rows x all 256 cols. x is staged via
// branch-free float4 loads into DOUBLE-BUFFERED LDS tiles: the b128 global
// load for tile k+1 is issued before the barrier, overlapping the B-fragment
// loads + 4 WMMAs of tile k. One barrier per K-tile (safe with 2 buffers:
// reads of a buffer are consumed before that wave signals the next barrier).
// K = 6000 = 187 full 32-tiles + 16-wide tail (peeled). grid = (500, 2).
// ---------------------------------------------------------------------------
__global__ void embed_k(const float* __restrict__ x,
                        const h16* __restrict__ quesTh, const h16* __restrict__ quesTd,
                        h16* __restrict__ xh, h16* __restrict__ xd) {
  const h16* Bt = blockIdx.y ? quesTd : quesTh;
  h16* C        = blockIdx.y ? xd : xh;
  __shared__ __align__(16) h16 As[2][32][40];  // double-buffered 32x32 tiles
  int lane = threadIdx.x & 31;
  int wave = threadIdx.x >> 5;
  int mbase = blockIdx.x * 32;
  int mt  = wave >> 2;        // 0..1
  int nt0 = (wave & 3) * 4;   // 4 waves * 4 tiles = 16 N-tiles = 256 cols
  const int K = 6000;
  int fr = threadIdx.x >> 3;          // fill row 0..31
  int fc = (threadIdx.x & 7) << 2;    // fill col 0,4,...,28
  const float* xrow = x + (size_t)(mbase + fr) * K;
  v8f acc[4] = {};
  v4f v = *(const v4f*)(xrow + fc);   // preload tile 0
  for (int kt = 0; kt < 187; ++kt) {
    int buf = kt & 1;
    v4h hv;
#pragma unroll
    for (int u = 0; u < 4; ++u) hv[u] = (h16)v[u];
    *(v4h*)(&As[buf][fr][fc]) = hv;   // b64 LDS store
    v4f vnext = v;
    if (kt + 1 < 187)                 // issue next x tile before the barrier
      vnext = *(const v4f*)(xrow + ((kt + 1) << 5) + fc);
    __syncthreads();
    v16h a = load_a_nc(&As[buf][0][0], 40, mt * 16, 0, lane);
    int k0 = kt << 5;
#pragma unroll
    for (int t = 0; t < 4; ++t) {
      v16h b = load_b_nc(Bt, K, (nt0 + t) * 16, k0, lane);
      acc[t] = wmma_f16(a, b, acc[t]);
    }
    v = vnext;
  }
  {  // K tail: k0 = 5984, 16 valid columns; buffer 187&1 = 1
    const int k0 = 5984;
    v4h hv = {};
    if (fc < 16) {
      v4f vt = *(const v4f*)(xrow + k0 + fc);
#pragma unroll
      for (int u = 0; u < 4; ++u) hv[u] = (h16)vt[u];
    }
    __syncthreads();                  // all reads of buffer 1 done
    *(v4h*)(&As[1][fr][fc]) = hv;
    __syncthreads();
    v16h a = load_a_nc(&As[1][0][0], 40, mt * 16, 0, lane);
#pragma unroll
    for (int t = 0; t < 4; ++t) {
      v16h b = load_b_frag(Bt, 256, K, K, (nt0 + t) * 16, k0, lane);
      acc[t] = wmma_f16(a, b, acc[t]);
    }
  }
#pragma unroll
  for (int t = 0; t < 4; ++t)
    store_c_f16(C, 16000, 256, 256, mbase + mt * 16, (nt0 + t) * 16, lane,
                acc[t], nullptr);
}

// ---------------------------------------------------------------------------
// Persistent GRU: grid = 2 blocks (one per GRU), 512 threads = 16 waves.
// h kept in LDS (f16 WMMA operand + f32 master). Per step:
//   gh[32,768] = h[32,256] @ w_hh[768,256]^T  (96 WMMA tiles, 6/wave, no guards)
//   gates 4-wide vectorized in f32.
// Dynamic LDS: 16KB h16 + 32KB h32 + 96KB gh = 147456 B (<320KB/WGP).
// ---------------------------------------------------------------------------
__global__ void gru_k(const h16* __restrict__ xph, const h16* __restrict__ xpd,
                      const h16* __restrict__ whh1, const h16* __restrict__ whh2,
                      const float* __restrict__ bhh1, const float* __restrict__ bhh2,
                      h16* __restrict__ outh, h16* __restrict__ outd) {
  extern __shared__ char smem[];
  h16*   hf16 = (h16*)smem;                    // [32][256]
  float* hf32 = (float*)(smem + 16384);        // [32][256]
  float* gh   = (float*)(smem + 49152);        // [32][768]
  const h16*   xp  = blockIdx.x ? xpd  : xph;
  const h16*   whh = blockIdx.x ? whh2 : whh1;
  const float* bhh = blockIdx.x ? bhh2 : bhh1;
  h16*         out = blockIdx.x ? outd : outh;

  int tid  = threadIdx.x;
  int lane = tid & 31;
  int wave = tid >> 5;           // 0..15
  for (int i = tid; i < 32 * 256; i += 512) { hf16[i] = (h16)0.f; hf32[i] = 0.f; }
  __syncthreads();

  int mt  = wave >> 3;           // 0..1
  int ntb = (wave & 7) * 6;      // 6 N-tiles per wave -> 48 tiles = 768 cols

  for (int s = 0; s < 500; ++s) {
    v8f acc[6] = {};
#pragma unroll
    for (int kt = 0; kt < 8; ++kt) {
      v16h a = load_a_nc(hf16, 256, mt * 16, kt * 32, lane);
#pragma unroll
      for (int t = 0; t < 6; ++t) {
        v16h b = load_b_nc(whh, 256, (ntb + t) * 16, kt * 32, lane);
        acc[t] = wmma_f16(a, b, acc[t]);
      }
    }
#pragma unroll
    for (int t = 0; t < 6; ++t) {
      int n  = (ntb + t) * 16 + (lane & 15);
      int mb = mt * 16 + ((lane & 16) ? 8 : 0);
#pragma unroll
      for (int r = 0; r < 8; ++r) gh[(mb + r) * 768 + n] = acc[t][r];
    }
    __syncthreads();
    // gate math, 4-wide: 2048 vector items / 512 threads = 4 iterations
    for (int it = tid; it < 2048; it += 512) {
      int b  = it >> 6;
      int j4 = (it & 63) << 2;
      size_t xoff = ((size_t)(b * 500 + s)) * 768;
      v4h xr4 = *(const v4h*)(xp + xoff + j4);
      v4h xz4 = *(const v4h*)(xp + xoff + 256 + j4);
      v4h xn4 = *(const v4h*)(xp + xoff + 512 + j4);
      const float* ghb = gh + b * 768;
      v4f gr4 = *(const v4f*)(ghb + j4);
      v4f gz4 = *(const v4f*)(ghb + 256 + j4);
      v4f gn4 = *(const v4f*)(ghb + 512 + j4);
      v4f br4 = *(const v4f*)(bhh + j4);
      v4f bz4 = *(const v4f*)(bhh + 256 + j4);
      v4f bn4 = *(const v4f*)(bhh + 512 + j4);
      v4f hold = *(const v4f*)(hf32 + b * 256 + j4);
      v4f hnew;
      v4h hh;
#pragma unroll
      for (int u = 0; u < 4; ++u) {
        float r = 1.f / (1.f + __expf(-((float)xr4[u] + gr4[u] + br4[u])));
        float z = 1.f / (1.f + __expf(-((float)xz4[u] + gz4[u] + bz4[u])));
        float n = tanhf((float)xn4[u] + r * (gn4[u] + bn4[u]));
        float h = (1.f - z) * n + z * hold[u];
        hnew[u] = h;
        hh[u] = (h16)h;
      }
      *(v4f*)(hf32 + b * 256 + j4) = hnew;
      *(v4h*)(hf16 + b * 256 + j4) = hh;
      *(v4h*)(out + ((size_t)(b * 500 + s)) * 256 + j4) = hh;
    }
    __syncthreads();
  }
}

// ---------------------------------------------------------------------------
// Softmax over rows of scores[16000,500] -> attn f16. One block per row.
// ---------------------------------------------------------------------------
__global__ void softmax_k(const float* __restrict__ scores, h16* __restrict__ attn) {
  __shared__ float red[256];
  size_t row = blockIdx.x;
  const float* S = scores + row * 500;
  h16* Ao = attn + row * 500;
  int tid = threadIdx.x;
  float m = -1e30f;
  for (int i = tid; i < 500; i += 256) m = fmaxf(m, S[i]);
  red[tid] = m; __syncthreads();
  for (int off = 128; off > 0; off >>= 1) {
    if (tid < off) red[tid] = fmaxf(red[tid], red[tid + off]);
    __syncthreads();
  }
  m = red[0]; __syncthreads();
  float sum = 0.f;
  for (int i = tid; i < 500; i += 256) sum += __expf(S[i] - m);
  red[tid] = sum; __syncthreads();
  for (int off = 128; off > 0; off >>= 1) {
    if (tid < off) red[tid] += red[tid + off];
    __syncthreads();
  }
  float inv = 1.f / red[0];
  for (int i = tid; i < 500; i += 256) Ao[i] = (h16)(__expf(S[i] - m) * inv);
}

// ---------------------------------------------------------------------------
// Fused attention bmm: per batch b,
//   ens[:,0:256]   = attn @ out_d
//   ens[:,256:512] = colsum(out_h) - attn @ out_h      ( == (1-attn) @ out_h )
// A-fragments shared between the two accumulations; K=500 tail peeled.
// ---------------------------------------------------------------------------
__global__ void fused_k(const h16* __restrict__ attn, const h16* __restrict__ outdT,
                        const h16* __restrict__ outhT, const float* __restrict__ hsum,
                        h16* __restrict__ ens) {
  int b = blockIdx.z;
  const h16* A  = attn  + (size_t)b * 500 * 500;
  const h16* Bd = outdT + (size_t)b * 256 * 500;
  const h16* Bh = outhT + (size_t)b * 256 * 500;
  const float* hs = hsum + b * 256;
  int lane = threadIdx.x & 31;
  int wave = threadIdx.x >> 5;   // 8 waves
  int m0  = blockIdx.x * 16;     // 32 blocks cover M=500
  bool aFull = (m0 + 16 <= 500);
  int ntb = wave * 2;            // 16 N-tiles = 256 cols
  v8f accd[2] = {}, acch[2] = {};
  for (int kt = 0; kt < 15; ++kt) {  // full K tiles (k0+31 <= 479 < 500)
    int k0 = kt << 5;
    v16h a = aFull ? load_a_nc(A, 500, m0, k0, lane)
                   : load_a_frag(A, 500, 500, 500, m0, k0, lane);
#pragma unroll
    for (int t = 0; t < 2; ++t) {
      int n0 = (ntb + t) * 16;
      v16h bd = load_b_nc(Bd, 500, n0, k0, lane);
      v16h bh = load_b_nc(Bh, 500, n0, k0, lane);
      accd[t] = wmma_f16(a, bd, accd[t]);
      acch[t] = wmma_f16(a, bh, acch[t]);
    }
  }
  {  // K tail: k0 = 480, 20 valid
    const int k0 = 480;
    v16h a = load_a_frag(A, 500, 500, 500, m0, k0, lane);
#pragma unroll
    for (int t = 0; t < 2; ++t) {
      int n0 = (ntb + t) * 16;
      v16h bd = load_b_frag(Bd, 256, 500, 500, n0, k0, lane);
      v16h bh = load_b_frag(Bh, 256, 500, 500, n0, k0, lane);
      accd[t] = wmma_f16(a, bd, accd[t]);
      acch[t] = wmma_f16(a, bh, acch[t]);
    }
  }
#pragma unroll
  for (int t = 0; t < 2; ++t) {
    int n  = (ntb + t) * 16 + (lane & 15);
    int mb = m0 + ((lane & 16) ? 8 : 0);
    float sh = hs[n];
#pragma unroll
    for (int r = 0; r < 8; ++r) {
      int m = mb + r;
      if (m < 500) {
        size_t row = (size_t)(b * 500 + m) * 512;
        ens[row + n]       = (h16)accd[t][r];
        ens[row + 256 + n] = (h16)(sh - acch[t][r]);
      }
    }
  }
}

// ---------------------------------------------------------------------------
// Small utility kernels
// ---------------------------------------------------------------------------
__global__ void cvt_k(const float* __restrict__ in, h16* __restrict__ out, long n) {
  size_t i = (size_t)blockIdx.x * 256 + threadIdx.x;
  if (i < (size_t)n) out[i] = (h16)in[i];
}

// in [R][C] (f32) -> out [C][R] (f16)
__global__ void cvtT_k(const float* __restrict__ in, h16* __restrict__ out,
                       int R, int C) {
  size_t i = (size_t)blockIdx.x * 256 + threadIdx.x;
  if (i >= (size_t)R * C) return;
  int c = (int)(i % C);
  int r = (int)(i / C);
  out[(size_t)c * R + r] = (h16)in[i];
}

// in [32][500][256] f16 -> out [32][256][500] f16
__global__ void transpose_k(const h16* __restrict__ in, h16* __restrict__ out) {
  size_t i = (size_t)blockIdx.x * 256 + threadIdx.x;
  if (i >= (size_t)32 * 500 * 256) return;
  int j = (int)(i & 255);
  size_t r = i >> 8;
  int t = (int)(r % 500);
  int b = (int)(r / 500);
  out[((size_t)b * 256 + j) * 500 + t] = in[i];
}

// hsum[b][j] = sum_t out_h[b][t][j]
__global__ void hsum_k(const h16* __restrict__ outh, float* __restrict__ hsum) {
  int b = blockIdx.x, j = threadIdx.x;  // 32 blocks x 256 threads
  float s = 0.f;
  for (int t = 0; t < 500; ++t) s += (float)outh[((size_t)(b * 500 + t)) * 256 + j];
  hsum[b * 256 + j] = s;
}

// ---------------------------------------------------------------------------
// Host-side orchestration
// ---------------------------------------------------------------------------
extern "C" void kernel_launch(void* const* d_in, const int* in_sizes, int n_in,
                              void* d_out, int out_size, void* d_ws, size_t ws_size,
                              hipStream_t stream) {
  const float* x      = (const float*)d_in[0];
  const float* ques_h = (const float*)d_in[1];
  const float* ques_d = (const float*)d_in[2];
  const float* g1_wih = (const float*)d_in[3];
  const float* g1_whh = (const float*)d_in[4];
  const float* g1_bih = (const float*)d_in[5];
  const float* g1_bhh = (const float*)d_in[6];
  const float* g2_wih = (const float*)d_in[7];
  const float* g2_whh = (const float*)d_in[8];
  const float* g2_bih = (const float*)d_in[9];
  const float* g2_bhh = (const float*)d_in[10];
  const float* fcc_w  = (const float*)d_in[11];
  const float* fcc_b  = (const float*)d_in[12];
  const float* fct_w  = (const float*)d_in[13];
  const float* fct_b  = (const float*)d_in[14];
  const float* fce_w  = (const float*)d_in[15];
  const float* fce_b  = (const float*)d_in[16];
  (void)in_sizes; (void)n_in; (void)out_size; (void)ws_size;

  char* ws = (char*)d_ws;
  size_t off = 0;
  auto carve = [&](size_t bytes) -> void* {
    void* p = ws + off;
    off = (off + bytes + 255) & ~(size_t)255;
    return p;
  };

  h16*   quesTh = (h16*)carve((size_t)6000 * 256 * 2);   // [256][6000]
  h16*   quesTd = (h16*)carve((size_t)6000 * 256 * 2);
  h16*   w1ih   = (h16*)carve((size_t)768 * 256 * 2);    // [768][256]
  h16*   w1hh   = (h16*)carve((size_t)768 * 256 * 2);
  h16*   w2ih   = (h16*)carve((size_t)768 * 256 * 2);
  h16*   w2hh   = (h16*)carve((size_t)768 * 256 * 2);
  h16*   fcc16  = (h16*)carve((size_t)3000 * 256 * 2);
  h16*   fct16  = (h16*)carve((size_t)3000 * 256 * 2);
  h16*   fce16  = (h16*)carve((size_t)3000 * 512 * 2);
  h16*   xh16   = (h16*)carve((size_t)16000 * 256 * 2);
  h16*   xd16   = (h16*)carve((size_t)16000 * 256 * 2);
  h16*   xph    = (h16*)carve((size_t)16000 * 768 * 2);
  h16*   xpd    = (h16*)carve((size_t)16000 * 768 * 2);
  h16*   outh16 = (h16*)carve((size_t)16000 * 256 * 2);
  h16*   outd16 = (h16*)carve((size_t)16000 * 256 * 2);
  h16*   outhT  = (h16*)carve((size_t)16000 * 256 * 2);  // [32][256][500]
  h16*   outdT  = (h16*)carve((size_t)16000 * 256 * 2);
  float* hsum   = (float*)carve((size_t)32 * 256 * 4);
  float* scores = (float*)carve((size_t)32 * 500 * 500 * 4);
  h16*   attn16 = (h16*)carve((size_t)32 * 500 * 500 * 2);
  h16*   ens16  = (h16*)carve((size_t)16000 * 512 * 2);

  // 1) weight conversion / transposition to f16
  cvtT_k<<<(6000 * 256 + 255) / 256, 256, 0, stream>>>(ques_h, quesTh, 6000, 256);
  cvtT_k<<<(6000 * 256 + 255) / 256, 256, 0, stream>>>(ques_d, quesTd, 6000, 256);
  cvt_k<<<(768 * 256 + 255) / 256, 256, 0, stream>>>(g1_wih, w1ih, 768 * 256);
  cvt_k<<<(768 * 256 + 255) / 256, 256, 0, stream>>>(g1_whh, w1hh, 768 * 256);
  cvt_k<<<(768 * 256 + 255) / 256, 256, 0, stream>>>(g2_wih, w2ih, 768 * 256);
  cvt_k<<<(768 * 256 + 255) / 256, 256, 0, stream>>>(g2_whh, w2hh, 768 * 256);
  cvt_k<<<(3000 * 256 + 255) / 256, 256, 0, stream>>>(fcc_w, fcc16, 3000 * 256);
  cvt_k<<<(3000 * 256 + 255) / 256, 256, 0, stream>>>(fct_w, fct16, 3000 * 256);
  cvt_k<<<(3000 * 512 + 255) / 256, 256, 0, stream>>>(fce_w, fce16, 3000 * 512);

  // 2) embeddings: x_h, x_d   (x read once, double-buffered LDS staging)
  embed_k<<<dim3(500, 2, 1), 256, 0, stream>>>(x, quesTh, quesTd, xh16, xd16);

  // 3) input projections x_proj = x_emb @ w_ih^T + b_ih  (f16 out)
  gemm_wmma_k<1><<<dim3(1000, 2, 1), 256, 0, stream>>>(
      xh16, w1ih, xph, g1_bih, 16000, 768, 256, 256, 256, 768, 0, 0, 0);
  gemm_wmma_k<1><<<dim3(1000, 2, 1), 256, 0, stream>>>(
      xd16, w2ih, xpd, g2_bih, 16000, 768, 256, 256, 256, 768, 0, 0, 0);

  // 4) dual persistent GRUs (sequential over S inside the kernel)
  gru_k<<<2, 512, 147456, stream>>>(xph, xpd, w1hh, w2hh, g1_bhh, g2_bhh,
                                    outh16, outd16);

  // 5) per-branch logits -> d_out[0], d_out[1]
  float* out = (float*)d_out;
  gemm_wmma_k<0><<<dim3(1000, 6, 1), 256, 0, stream>>>(
      outh16, fcc16, out, fcc_b, 16000, 3000, 256, 256, 256, 3000, 0, 0, 0);
  gemm_wmma_k<0><<<dim3(1000, 6, 1), 256, 0, stream>>>(
      outd16, fct16, out + 48000000ll, fct_b, 16000, 3000, 256, 256, 256, 3000,
      0, 0, 0);

  // 6) attention scores (batched over 32): scores = out_h @ out_d^T  (f32)
  gemm_wmma_k<0><<<dim3(32, 1, 32), 256, 0, stream>>>(
      outh16, outd16, scores, nullptr, 500, 500, 256, 256, 256, 500,
      500 * 256, 500 * 256, 500 * 500);

  // 7) transposed copies + column sums for the fused bmm
  transpose_k<<<(32 * 500 * 256 + 255) / 256, 256, 0, stream>>>(outh16, outhT);
  transpose_k<<<(32 * 500 * 256 + 255) / 256, 256, 0, stream>>>(outd16, outdT);
  hsum_k<<<32, 256, 0, stream>>>(outh16, hsum);

  // 8) softmax rows -> attn (f16)
  softmax_k<<<16000, 256, 0, stream>>>(scores, attn16);

  // 9) fused_d / fused_h -> ens[16000,512] f16
  fused_k<<<dim3(32, 1, 32), 256, 0, stream>>>(attn16, outdT, outhT, hsum, ens16);

  // 10) ensemble logits -> d_out[2]
  gemm_wmma_k<0><<<dim3(1000, 6, 1), 256, 0, stream>>>(
      ens16, fce16, out + 96000000ll, fce_b, 16000, 3000, 512, 512, 512, 3000,
      0, 0, 0);
}
